// TrafalignTransformer_66013647339884
// MI455X (gfx1250) — matrix operand order repaired and checked by reading, not compiled, not run on hardware
//
#include <hip/hip_runtime.h>
#include <hip/hip_bf16.h>
#include <stdint.h>

// ---------------------------------------------------------------------------
// Types for CDNA5 WMMA
// ---------------------------------------------------------------------------
typedef __bf16 bf16_t;
typedef __attribute__((ext_vector_type(16))) __bf16 v16bf;
typedef __attribute__((ext_vector_type(8)))  __bf16 v8bf;
typedef __attribute__((ext_vector_type(8)))  float  v8f;

// Problem constants (from reference)
#define BB   2
#define HH   128
#define WW   128
#define CC   256
#define DD   64
#define PP   18
#define NN   (HH * WW)            // 16384
#define MM   (BB * NN)            // 32768 rows
#define C4   (4 * CC)             // 1024
#define OUT2_ELEMS ((size_t)BB * NN * CC)   // 8,388,608

// ---------------------------------------------------------------------------
// Workspace layout (bytes, 256-aligned regions)
// ---------------------------------------------------------------------------
#define OFF_XPBF   ((size_t)0)                         // bf16 [M,256]  (reused as Xn_bf16)
#define OFF_QKV    ((size_t)16777216)                  // f32  [M,192]
#define OFF_CTX    ((size_t)41943040)                  // bf16 [M,64]
#define OFF_AOUT   ((size_t)46137344)                  // f32  [M,256]  (attn-out, reused as ffn-out)
#define OFF_XN     ((size_t)79691776)                  // f32  [M,256]
#define OFF_HBF    ((size_t)113246208)                 // bf16 [M,1024]
#define OFF_WQKVT  ((size_t)180355072)                 // bf16 [192,256]
#define OFF_WOT    (OFF_WQKVT + 98304)                 // bf16 [256,64]
#define OFF_W1T    (OFF_WOT   + 32768)                 // bf16 [1024,256]
#define OFF_W2T    (OFF_W1T   + 524288)                // bf16 [256,1024]
#define OFF_BQKV   (OFF_W2T   + 524288)                // f32  [192]

// ---------------------------------------------------------------------------
// 1) Xp_bf16 = bf16(X + pos_enc)  (pos broadcast over batch; N*C = 2^22)
// ---------------------------------------------------------------------------
__global__ __launch_bounds__(256)
void prep_kernel(const float* __restrict__ X, const float* __restrict__ pos,
                 bf16_t* __restrict__ xpbf)
{
    size_t t = (size_t)blockIdx.x * 256 + threadIdx.x;
    size_t e = t * 4;
    if (e >= (size_t)MM * CC) return;
    const float4 x = *(const float4*)(X + e);
    const float4 p = *(const float4*)(pos + (e & ((size_t)NN * CC - 1)));
    union { bf16_t h[4]; uint2 u; } pk;
    pk.h[0] = (bf16_t)(x.x + p.x);
    pk.h[1] = (bf16_t)(x.y + p.y);
    pk.h[2] = (bf16_t)(x.z + p.z);
    pk.h[3] = (bf16_t)(x.w + p.w);
    *(uint2*)(xpbf + e) = pk.u;
}

// ---------------------------------------------------------------------------
// 2) Weight transpose + f32->bf16:  src[R,Cc] row-major -> dst[Cc,R]
// ---------------------------------------------------------------------------
__global__ __launch_bounds__(256)
void transpose_cvt_kernel(const float* __restrict__ src, bf16_t* __restrict__ dst,
                          int R, int Cc)
{
    int idx = blockIdx.x * 256 + threadIdx.x;
    if (idx >= R * Cc) return;
    int r = idx / Cc;
    int c = idx - r * Cc;
    dst[(size_t)c * R + r] = (bf16_t)src[idx];
}

__global__ void concat3_kernel(const float* __restrict__ a, const float* __restrict__ b,
                               const float* __restrict__ c, float* __restrict__ dst, int n)
{
    int i = threadIdx.x;
    if (i < n)           dst[i] = a[i];
    else if (i < 2 * n)  dst[i] = b[i - n];
    else if (i < 3 * n)  dst[i] = c[i - 2 * n];
}

// ---------------------------------------------------------------------------
// bf16 GEMM:  out[M,N] = act( A[M,K] @ Wt[N,K]^T + bias )
// One wave per 32(M) x 64(N) output tile:
//   2 A-fragments x 4 B-fragments -> 8 f32 accumulators, 8 WMMAs / k-step.
// A layout (16-bit 16x32): lane m=l&15; kh=l>>4; elems e -> K=kh*16+e.
// B layout (16-bit 32x16): lane n=l&15; kh=l>>4; elems e -> K=kh*16+e
//   (contiguous after the [N,K] weight transpose).
// C/D layout: lane n=l&15; VGPR r -> row r + (l>>4)*8.
// ---------------------------------------------------------------------------
#define WMMA_BF16(A_, B_, C_) \
    __builtin_amdgcn_wmma_f32_16x16x32_bf16(false, (A_), false, (B_), (short)0, (C_), false, false)

template<bool RELU, bool OUT_BF16>
__global__ __launch_bounds__(256)
void gemm_bf16_kernel(const bf16_t* __restrict__ A, const bf16_t* __restrict__ Wt,
                      const float* __restrict__ bias, void* __restrict__ outv,
                      int M, int N, int K)
{
    const int lane   = threadIdx.x & 31;
    const int wv     = threadIdx.x >> 5;
    const int tilesM = M >> 5;           // 32-row tiles
    const int tile   = blockIdx.x * 8 + wv;
    const int tm     = tile % tilesM;
    const int tn     = tile / tilesM;
    const int n0     = tn * 64;
    if (n0 >= N) return;                 // uniform per wave

    const int m  = lane & 15;
    const int kh = lane >> 4;

    const bf16_t* Arow0 = A + (size_t)(tm * 32 + m) * K;
    const bf16_t* Arow1 = Arow0 + (size_t)16 * K;
    const bf16_t* B0    = Wt + (size_t)(n0 + m) * K + kh * 16;
    const size_t  bstep = (size_t)16 * K;

    v8f acc00 = {}, acc01 = {}, acc02 = {}, acc03 = {};
    v8f acc10 = {}, acc11 = {}, acc12 = {}, acc13 = {};

    for (int k0 = 0; k0 < K; k0 += 32) {
        v8bf a0lo = *(const v8bf*)(Arow0 + k0 + kh * 16);
        v8bf a0hi = *(const v8bf*)(Arow0 + k0 + 8 + kh * 16);
        v8bf a1lo = *(const v8bf*)(Arow1 + k0 + kh * 16);
        v8bf a1hi = *(const v8bf*)(Arow1 + k0 + 8 + kh * 16);
        v16bf a0 = __builtin_shufflevector(a0lo, a0hi,
                     0, 1, 2, 3, 4, 5, 6, 7, 8, 9, 10, 11, 12, 13, 14, 15);
        v16bf a1 = __builtin_shufflevector(a1lo, a1hi,
                     0, 1, 2, 3, 4, 5, 6, 7, 8, 9, 10, 11, 12, 13, 14, 15);

        v16bf b0 = *(const v16bf*)(B0 + k0);
        v16bf b1 = *(const v16bf*)(B0 + bstep + k0);
        v16bf b2 = *(const v16bf*)(B0 + 2 * bstep + k0);
        v16bf b3 = *(const v16bf*)(B0 + 3 * bstep + k0);

        acc00 = WMMA_BF16(a0, b0, acc00);
        acc10 = WMMA_BF16(a1, b0, acc10);
        acc01 = WMMA_BF16(a0, b1, acc01);
        acc11 = WMMA_BF16(a1, b1, acc11);
        acc02 = WMMA_BF16(a0, b2, acc02);
        acc12 = WMMA_BF16(a1, b2, acc12);
        acc03 = WMMA_BF16(a0, b3, acc03);
        acc13 = WMMA_BF16(a1, b3, acc13);
    }

    const float bv0 = bias[n0 + m];
    const float bv1 = bias[n0 + 16 + m];
    const float bv2 = bias[n0 + 32 + m];
    const float bv3 = bias[n0 + 48 + m];

    const int mbase0 = tm * 32 + kh * 8;        // rows 0-15 half
    const int mbase1 = mbase0 + 16;             // rows 16-31 half

    #pragma unroll
    for (int r = 0; r < 8; ++r) {
        float o00 = acc00[r] + bv0, o01 = acc01[r] + bv1;
        float o02 = acc02[r] + bv2, o03 = acc03[r] + bv3;
        float o10 = acc10[r] + bv0, o11 = acc11[r] + bv1;
        float o12 = acc12[r] + bv2, o13 = acc13[r] + bv3;
        if (RELU) {
            o00 = fmaxf(o00, 0.0f); o01 = fmaxf(o01, 0.0f);
            o02 = fmaxf(o02, 0.0f); o03 = fmaxf(o03, 0.0f);
            o10 = fmaxf(o10, 0.0f); o11 = fmaxf(o11, 0.0f);
            o12 = fmaxf(o12, 0.0f); o13 = fmaxf(o13, 0.0f);
        }
        if (OUT_BF16) {
            bf16_t* p0 = (bf16_t*)outv + (size_t)(mbase0 + r) * N + n0 + m;
            bf16_t* p1 = (bf16_t*)outv + (size_t)(mbase1 + r) * N + n0 + m;
            p0[0]  = (bf16_t)o00; p0[16] = (bf16_t)o01;
            p0[32] = (bf16_t)o02; p0[48] = (bf16_t)o03;
            p1[0]  = (bf16_t)o10; p1[16] = (bf16_t)o11;
            p1[32] = (bf16_t)o12; p1[48] = (bf16_t)o13;
        } else {
            float* p0 = (float*)outv + (size_t)(mbase0 + r) * N + n0 + m;
            float* p1 = (float*)outv + (size_t)(mbase1 + r) * N + n0 + m;
            p0[0] = o00; p0[16] = o01; p0[32] = o02; p0[48] = o03;
            p1[0] = o10; p1[16] = o11; p1[32] = o12; p1[48] = o13;
        }
    }
}

// ---------------------------------------------------------------------------
// 3) Random-point attention: one wave32 per query pixel.
//    QKV layout: f32 [M,192] = (Q|K|V). Lane owns 2 of 64 dims.
// ---------------------------------------------------------------------------
__global__ __launch_bounds__(256)
void attn_kernel(const float* __restrict__ QKV, const int* __restrict__ sel,
                 float* __restrict__ attn_out, bf16_t* __restrict__ ctx)
{
    const int lane = threadIdx.x & 31;
    const int wv   = threadIdx.x >> 5;
    const int q    = blockIdx.x * 8 + wv;      // 0..M-1
    const int b    = q >> 14;                  // N = 16384 = 2^14
    const int i    = q & (NN - 1);

    const float* qkvb = QKV + (size_t)b * NN * 192;
    const float2 qv = *(const float2*)(qkvb + (size_t)i * 192 + lane * 2);
    const int* idxp = sel + (size_t)q * PP;

    int   jidx[PP];
    float sc[PP];
    #pragma unroll
    for (int p = 0; p < PP; ++p) {
        const int j = idxp[p];
        jidx[p] = j;
        const float2 kv = *(const float2*)(qkvb + (size_t)j * 192 + 64 + lane * 2);
        float d = qv.x * kv.x + qv.y * kv.y;
        #pragma unroll
        for (int off = 16; off; off >>= 1) d += __shfl_xor(d, off, 32);
        sc[p] = d * 0.125f;                    // 1/sqrt(64)
    }

    float mx = sc[0];
    #pragma unroll
    for (int p = 1; p < PP; ++p) mx = fmaxf(mx, sc[p]);
    float sum = 0.0f;
    #pragma unroll
    for (int p = 0; p < PP; ++p) { float e = __expf(sc[p] - mx); sc[p] = e; sum += e; }
    const float inv = 1.0f / sum;

    if (lane < PP) attn_out[(size_t)q * PP + lane] = sc[lane] * inv;

    float cx = 0.0f, cy = 0.0f;
    #pragma unroll
    for (int p = 0; p < PP; ++p) {
        const float a = sc[p] * inv;
        const float2 vv = *(const float2*)(qkvb + (size_t)jidx[p] * 192 + 128 + lane * 2);
        cx += a * vv.x;
        cy += a * vv.y;
    }
    union { bf16_t h[2]; unsigned u; } pk;
    pk.h[0] = (bf16_t)cx;
    pk.h[1] = (bf16_t)cy;
    *(unsigned*)(ctx + (size_t)q * DD + lane * 2) = pk.u;
}

// ---------------------------------------------------------------------------
// LayerNorm over C=256, one wave per row; lane owns 8 channels.
// ADD_POS: x = base + pos + delta (LN1, recomputes Xp); else x = base + delta.
// ---------------------------------------------------------------------------
template<bool ADD_POS, bool WRITE_BF16>
__global__ __launch_bounds__(256)
void ln_kernel(const float* __restrict__ base, const float* __restrict__ pos,
               const float* __restrict__ delta, const float* __restrict__ gamma,
               const float* __restrict__ beta, float* __restrict__ out_f32,
               bf16_t* __restrict__ out_bf16)
{
    const int lane = threadIdx.x & 31;
    const int wv   = threadIdx.x >> 5;
    const int row  = blockIdx.x * 8 + wv;
    const size_t rb = (size_t)row * CC;
    const int c0 = lane * 8;

    float x[8];
    {
        float4 a0 = *(const float4*)(base + rb + c0);
        float4 a1 = *(const float4*)(base + rb + c0 + 4);
        float4 d0 = *(const float4*)(delta + rb + c0);
        float4 d1 = *(const float4*)(delta + rb + c0 + 4);
        x[0] = a0.x + d0.x; x[1] = a0.y + d0.y; x[2] = a0.z + d0.z; x[3] = a0.w + d0.w;
        x[4] = a1.x + d1.x; x[5] = a1.y + d1.y; x[6] = a1.z + d1.z; x[7] = a1.w + d1.w;
        if (ADD_POS) {
            const size_t pb = (size_t)(row & (NN - 1)) * CC;
            float4 p0 = *(const float4*)(pos + pb + c0);
            float4 p1 = *(const float4*)(pos + pb + c0 + 4);
            x[0] += p0.x; x[1] += p0.y; x[2] += p0.z; x[3] += p0.w;
            x[4] += p1.x; x[5] += p1.y; x[6] += p1.z; x[7] += p1.w;
        }
    }

    float s = 0.0f, s2 = 0.0f;
    #pragma unroll
    for (int k = 0; k < 8; ++k) { s += x[k]; s2 += x[k] * x[k]; }
    #pragma unroll
    for (int off = 16; off; off >>= 1) {
        s  += __shfl_xor(s,  off, 32);
        s2 += __shfl_xor(s2, off, 32);
    }
    const float mean = s * (1.0f / CC);
    const float var  = s2 * (1.0f / CC) - mean * mean;
    const float rs   = rsqrtf(var + 1e-5f);

    #pragma unroll
    for (int k = 0; k < 8; ++k) {
        const float o = (x[k] - mean) * rs * gamma[c0 + k] + beta[c0 + k];
        out_f32[rb + c0 + k] = o;
        if (WRITE_BF16) out_bf16[rb + c0 + k] = (bf16_t)o;
    }
}

// ---------------------------------------------------------------------------
// Launch
// ---------------------------------------------------------------------------
extern "C" void kernel_launch(void* const* d_in, const int* in_sizes, int n_in,
                              void* d_out, int out_size, void* d_ws, size_t ws_size,
                              hipStream_t stream)
{
    const float* X   = (const float*)d_in[0];
    const int*   sel = (const int*)  d_in[1];
    const float* pos = (const float*)d_in[3];
    const float* Wq  = (const float*)d_in[4];
    const float* bq  = (const float*)d_in[5];
    const float* Wk  = (const float*)d_in[6];
    const float* bk  = (const float*)d_in[7];
    const float* Wv  = (const float*)d_in[8];
    const float* bv  = (const float*)d_in[9];
    const float* Wo  = (const float*)d_in[10];
    const float* bo  = (const float*)d_in[11];
    const float* W1  = (const float*)d_in[12];
    const float* b1  = (const float*)d_in[13];
    const float* W2  = (const float*)d_in[14];
    const float* b2  = (const float*)d_in[15];
    const float* g1  = (const float*)d_in[16];
    const float* be1 = (const float*)d_in[17];
    const float* g2  = (const float*)d_in[18];
    const float* be2 = (const float*)d_in[19];

    char* ws = (char*)d_ws;
    bf16_t* XPBF  = (bf16_t*)(ws + OFF_XPBF);     // also Xn_bf16
    float*  QKV   = (float*) (ws + OFF_QKV);
    bf16_t* CTX   = (bf16_t*)(ws + OFF_CTX);
    float*  AOUT  = (float*) (ws + OFF_AOUT);     // attn-out, then ffn-out
    float*  XN    = (float*) (ws + OFF_XN);
    bf16_t* HBF   = (bf16_t*)(ws + OFF_HBF);
    bf16_t* WQKVT = (bf16_t*)(ws + OFF_WQKVT);
    bf16_t* WOT   = (bf16_t*)(ws + OFF_WOT);
    bf16_t* W1T   = (bf16_t*)(ws + OFF_W1T);
    bf16_t* W2T   = (bf16_t*)(ws + OFF_W2T);
    float*  BQKV  = (float*) (ws + OFF_BQKV);

    float* out2_f32 = (float*)d_out;
    float* attn_f32 = (float*)d_out + OUT2_ELEMS;

    // 1) Xp in bf16
    prep_kernel<<<16384, 256, 0, stream>>>(X, pos, XPBF);

    // 2) weights -> bf16 [N,K]
    transpose_cvt_kernel<<<64,   256, 0, stream>>>(Wq, WQKVT,             CC, DD);
    transpose_cvt_kernel<<<64,   256, 0, stream>>>(Wk, WQKVT + 64  * CC,  CC, DD);
    transpose_cvt_kernel<<<64,   256, 0, stream>>>(Wv, WQKVT + 128 * CC,  CC, DD);
    transpose_cvt_kernel<<<64,   256, 0, stream>>>(Wo, WOT,               DD, CC);
    transpose_cvt_kernel<<<1024, 256, 0, stream>>>(W1, W1T,               CC, C4);
    transpose_cvt_kernel<<<1024, 256, 0, stream>>>(W2, W2T,               C4, CC);
    concat3_kernel<<<1, 192, 0, stream>>>(bq, bk, bv, BQKV, DD);

    // 3) fused QKV GEMM: [M,256] x [256,192] -> f32 [M,192]   (1024*3/8 blocks)
    gemm_bf16_kernel<false, false><<<384, 256, 0, stream>>>(XPBF, WQKVT, BQKV, QKV, MM, 192, CC);

    // 4) attention -> attn (d_out tail) + ctx bf16
    attn_kernel<<<4096, 256, 0, stream>>>(QKV, sel, attn_f32, CTX);

    // 5) out-proj: [M,64] x [64,256] -> f32 [M,256]           (1024*4/8 blocks)
    gemm_bf16_kernel<false, false><<<512, 256, 0, stream>>>(CTX, WOT, bo, AOUT, MM, CC, DD);

    // 6) LN1: Xn = LN(X + pos + attn_out), write f32 + bf16
    ln_kernel<true, true><<<4096, 256, 0, stream>>>(X, pos, AOUT, g1, be1, XN, XPBF);

    // 7) FFN1: relu([M,256] x [256,1024]) -> bf16 [M,1024]    (1024*16/8 blocks)
    gemm_bf16_kernel<true, true><<<2048, 256, 0, stream>>>(XPBF, W1T, b1, HBF, MM, C4, CC);

    // 8) FFN2: [M,1024] x [1024,256] -> f32 [M,256]           (1024*4/8 blocks)
    gemm_bf16_kernel<false, false><<<512, 256, 0, stream>>>(HBF, W2T, b2, AOUT, MM, CC, C4);

    // 9) LN2 -> d_out out2 region
    ln_kernel<false, false><<<4096, 256, 0, stream>>>(XN, nullptr, AOUT, g2, be2, out2_f32, nullptr);
}
